// GAT_88802743812680
// MI455X (gfx1250) — matrix operand order repaired and checked by reading, not compiled, not run on hardware
//
#include <hip/hip_runtime.h>
#include <cstdint>
#include <cstddef>

typedef __attribute__((ext_vector_type(16))) _Float16 v16h;
typedef __attribute__((ext_vector_type(8)))  float    v8f;

// ---------------------------------------------------------------------------
// Dense  Y = tanh(X @ W.T + b)  via v_wmma_f32_16x16x32_f16
//   X: [Nn, Din] f32 row-major, W: [Dout, Din] f32 row-major, b: [Dout]
//   One wave -> 16x16 f32 tile; block = (Dout/16) waves covering 16 rows.
// ---------------------------------------------------------------------------
template<int Din, int Dout>
__global__ __launch_bounds__(32 * (Dout / 16))
void gat_gemm_bias_tanh_wmma(const float* __restrict__ X,
                             const float* __restrict__ W,
                             const float* __restrict__ bias,
                             float* __restrict__ Y,
                             int Nn)
{
    const int lane = threadIdx.x & 31;
    const int wave = threadIdx.x >> 5;      // output-column tile
    const int tile = blockIdx.x;            // node tile (16 rows)
    const int nidx = lane & 15;             // M for A-frag, N for B/C/D
    const int half = lane >> 4;

    int row = tile * 16 + nidx;
    if (row >= Nn) row = Nn - 1;            // clamp: keep EXEC all-ones for WMMA
    const int col = wave * 16 + nidx;       // output feature

    const float* __restrict__ xr = X + (size_t)row * Din;
    const float* __restrict__ wr = W + (size_t)col * Din;

    v8f acc = {};
#pragma unroll
    for (int k = 0; k < Din; k += 32) {
        v16h a, b;
        // A 16x32 f16 layout: half 0 -> K {k..k+7, k+16..k+23}; half 1 -> +8
#pragma unroll
        for (int j = 0; j < 8; ++j) {
            a[j]     = (_Float16)xr[k + half * 8 + j];
            a[j + 8] = (_Float16)xr[k + 16 + half * 8 + j];
        }
        // B 32x16 f16 layout: lane holds column col; half 0 -> K {k..k+7, k+16..k+23}; half 1 -> +8
#pragma unroll
        for (int j = 0; j < 16; ++j) {
            b[j] = (_Float16)wr[k + half * 8 + (j & 7) + (j >> 3) * 16];
        }
        acc = __builtin_amdgcn_wmma_f32_16x16x32_f16(
            false, a, false, b, (short)0, acc, false, false);
    }

    const float bc = bias[col];
    // C/D layout: VGPR r -> M = r + 8*half, N = lane%16
#pragma unroll
    for (int r = 0; r < 8; ++r) {
        const int orow = tile * 16 + half * 8 + r;
        if (orow < Nn)
            Y[(size_t)orow * Dout + col] = tanhf(acc[r] + bc);
    }
}

// ---------------------------------------------------------------------------
// Per-edge logit + softmax numerator.  e = tanh(dot(a[s]+a[d], we) + be).
// tanh output in (-1,1) -> exp(e) is numerically safe without segment-max.
// ---------------------------------------------------------------------------
__global__ __launch_bounds__(256)
void gat_edge_logit(const float* __restrict__ A,      // [Nn, 64]
                    const int* __restrict__ src,
                    const int* __restrict__ dst,
                    const float* __restrict__ We,     // [64]
                    const float* __restrict__ be,     // [1]
                    float* __restrict__ exbuf,        // [E]
                    float* __restrict__ denom,        // [Nn] (zeroed)
                    int E)
{
    const int i = blockIdx.x * blockDim.x + threadIdx.x;
    if (i >= E) return;
    const int s = src[i], d = dst[i];
    const float4* __restrict__ as = (const float4*)(A + (size_t)s * 64);
    const float4* __restrict__ ad = (const float4*)(A + (size_t)d * 64);
    const float4* __restrict__ w4 = (const float4*)We;
    float e = 0.f;
#pragma unroll
    for (int k = 0; k < 16; ++k) {
        const float4 x = as[k], y = ad[k], w = w4[k];
        e += (x.x + y.x) * w.x + (x.y + y.y) * w.y +
             (x.z + y.z) * w.z + (x.w + y.w) * w.w;
    }
    const float ex = expf(tanhf(e + be[0]));
    exbuf[i] = ex;
    atomicAdd(&denom[s], ex);
}

// ---------------------------------------------------------------------------
// Attention-weighted scatter:  agg[s] += (ex/denom[s]) * v[d]
// ---------------------------------------------------------------------------
__global__ __launch_bounds__(256)
void gat_edge_scatter(const float* __restrict__ V,      // [Nn, 64]
                      const int* __restrict__ src,
                      const int* __restrict__ dst,
                      const float* __restrict__ exbuf,
                      const float* __restrict__ denom,
                      float* __restrict__ agg,          // [Nn, 64] (zeroed)
                      int E)
{
    const int i = blockIdx.x * blockDim.x + threadIdx.x;
    if (i >= E) return;
    const int s = src[i], d = dst[i];
    const float att = exbuf[i] / denom[s];
    const float4* __restrict__ vd = (const float4*)(V + (size_t)d * 64);
    float* __restrict__ out = agg + (size_t)s * 64;
#pragma unroll
    for (int k = 0; k < 16; ++k) {
        const float4 vv = vd[k];
        atomicAdd(&out[4 * k + 0], att * vv.x);
        atomicAdd(&out[4 * k + 1], att * vv.y);
        atomicAdd(&out[4 * k + 2], att * vv.z);
        atomicAdd(&out[4 * k + 3], att * vv.w);
    }
}

// ---------------------------------------------------------------------------
// Row normalize by unbiased std (torch.std ddof=1): y = x / std(x)
// ---------------------------------------------------------------------------
__global__ __launch_bounds__(256)
void gat_row_norm(const float* __restrict__ X, float* __restrict__ Y, int Nn)
{
    const int i = blockIdx.x * blockDim.x + threadIdx.x;
    if (i >= Nn) return;
    const float* __restrict__ r = X + (size_t)i * 64;
    float sum = 0.f, sumsq = 0.f;
#pragma unroll 8
    for (int k = 0; k < 64; ++k) {
        const float x = r[k];
        sum += x;
        sumsq += x * x;
    }
    const float mean = sum * (1.0f / 64.0f);
    const float var = (sumsq - 64.0f * mean * mean) * (1.0f / 63.0f);
    const float inv = rsqrtf(var);
#pragma unroll 8
    for (int k = 0; k < 64; ++k)
        Y[(size_t)i * 64 + k] = r[k] * inv;
}

__global__ __launch_bounds__(256)
void gat_zero_f32(float* __restrict__ p, size_t n)
{
    const size_t i = (size_t)blockIdx.x * blockDim.x + threadIdx.x;
    if (i < n) p[i] = 0.f;
}

// ---------------------------------------------------------------------------
extern "C" void kernel_launch(void* const* d_in, const int* in_sizes, int n_in,
                              void* d_out, int out_size, void* d_ws, size_t ws_size,
                              hipStream_t stream)
{
    (void)n_in; (void)out_size; (void)ws_size;

    const float* h   = (const float*)d_in[0];
    const int*   ei  = (const int*)d_in[1];
    const float* W11 = (const float*)d_in[2];
    const float* b11 = (const float*)d_in[3];
    const float* W12 = (const float*)d_in[4];
    const float* b12 = (const float*)d_in[5];
    const float* W13 = (const float*)d_in[6];
    const float* b13 = (const float*)d_in[7];
    const float* W21 = (const float*)d_in[8];
    const float* b21 = (const float*)d_in[9];
    const float* W22 = (const float*)d_in[10];
    const float* b22 = (const float*)d_in[11];
    const float* W23 = (const float*)d_in[12];
    const float* b23 = (const float*)d_in[13];

    const int Nn = in_sizes[0] / 128;   // 100,000
    const int E  = in_sizes[1] / 2;     // 1,600,000
    const int* src = ei;
    const int* dst = ei + E;

    // Workspace carve (agg and denom adjacent -> single zero-fill per layer)
    float* a     = (float*)d_ws;                    // [Nn,64]
    float* v     = a   + (size_t)Nn * 64;           // [Nn,64]
    float* h1    = v   + (size_t)Nn * 64;           // [Nn,64]
    float* exb   = h1  + (size_t)Nn * 64;           // [E]
    float* agg   = exb + (size_t)E;                 // [Nn,64]
    float* denom = agg + (size_t)Nn * 64;           // [Nn]

    const size_t zcount = (size_t)Nn * 64 + (size_t)Nn;   // agg + denom
    const int zgrid = (int)((zcount + 255) / 256);
    const int egrid = (E + 255) / 256;
    const int ngrid = (Nn + 255) / 256;
    const int tgrid = (Nn + 15) / 16;

    // ---------------- layer 1 (Din=128) ----------------
    gat_zero_f32<<<zgrid, 256, 0, stream>>>(agg, zcount);
    gat_gemm_bias_tanh_wmma<128, 64><<<tgrid, 128, 0, stream>>>(h, W11, b11, a, Nn);
    gat_gemm_bias_tanh_wmma<128, 64><<<tgrid, 128, 0, stream>>>(h, W13, b13, v, Nn);
    gat_edge_logit  <<<egrid, 256, 0, stream>>>(a, src, dst, W12, b12, exb, denom, E);
    gat_edge_scatter<<<egrid, 256, 0, stream>>>(v, src, dst, exb, denom, agg, E);
    gat_row_norm    <<<ngrid, 256, 0, stream>>>(agg, h1, Nn);

    // ---------------- layer 2 (Din=64) -----------------
    gat_zero_f32<<<zgrid, 256, 0, stream>>>(agg, zcount);
    gat_gemm_bias_tanh_wmma<64, 64><<<tgrid, 128, 0, stream>>>(h1, W21, b21, a, Nn);
    gat_gemm_bias_tanh_wmma<64, 64><<<tgrid, 128, 0, stream>>>(h1, W23, b23, v, Nn);
    gat_edge_logit  <<<egrid, 256, 0, stream>>>(a, src, dst, W22, b22, exb, denom, E);
    gat_edge_scatter<<<egrid, 256, 0, stream>>>(v, src, dst, exb, denom, agg, E);
    gat_row_norm    <<<ngrid, 256, 0, stream>>>(agg, (float*)d_out, Nn);
}
